// DynamicVAEDecoder_59047210385599
// MI455X (gfx1250) — matrix operand order repaired and checked by reading, not compiled, not run on hardware
//
#include <hip/hip_runtime.h>

// ---------------------------------------------------------------------------
// DynamicVAEDecoder for MI455X (gfx1250, wave32, WMMA bf16)
//
// reconNode [100000,119], reconEdgeType [400000,4], reconEdgeDir [400000,3]
// All GEMMs run through v_wmma_f32_16x16x32_bf16 with f32 accumulation.
// 64-row M tiles (4 strips/wave) + persistent accumulators: each B fragment
// (global, L2-hot) feeds 4 WMMAs; each A fragment (LDS) feeds NTW WMMAs.
// Uses CDNA5 320KB/WGP LDS: node WG = 86KB, edge WG = 123KB.
// ---------------------------------------------------------------------------

typedef __attribute__((ext_vector_type(16))) __bf16 v16bf;
typedef __attribute__((ext_vector_type(8)))  float  v8f;
typedef unsigned int  u32;
typedef unsigned short u16;

union BF16Frag { v16bf v; u32 u[8]; };

static constexpr int N_NODES = 100000;
static constexpr int N_EDGES = 400000;
static constexpr int KP_NODE = 288;   // 264 padded to 9 k-steps of 32
static constexpr int KP_EDGE = 576;   // 2 x 288
static constexpr long OUT_T  = 11900000L;  // 100000*119
static constexpr long OUT_D  = 13500000L;  // + 400000*4

__device__ __forceinline__ u16 f2bf(float f) {
  u32 u = __float_as_uint(f);
  u32 r = u + 0x7FFFu + ((u >> 16) & 1u);   // round-to-nearest-even
  return (u16)(r >> 16);
}

// ---------------------------------------------------------------------------
// Weight prep: f32 [srcK, srcN] -> bf16, zero-padded, pre-swizzled into the
// k-contiguous B-fragment order used here: dword index = ((kt*ntiles + nt)*32 + lane)*8 + v,
// lane 0-15: N = nt*16+lane, K = kt*32 + 2v(+1); lane 16-31: K += 16.
// split==1: edge W1 row remap (rows 0..263 -> 0..287 pad, 264..527 -> 288..575)
// ---------------------------------------------------------------------------
__device__ __forceinline__ u16 fetch_bf(const float* W, int k, int n,
                                        int srcK, int srcN, int split) {
  int sk;
  if (split) {
    sk = (k < 288) ? (k < 264 ? k : -1)
                   : ((k - 288) < 264 ? (k - 288) + 264 : -1);
  } else {
    sk = (k < srcK) ? k : -1;
  }
  if (sk < 0 || n >= srcN) return 0;
  return f2bf(W[(long)sk * srcN + n]);
}

__global__ void prep_wswz(const float* __restrict__ W, u32* __restrict__ dst,
                          int srcK, int srcN, int ntiles, int split, int total) {
  int idx = blockIdx.x * blockDim.x + threadIdx.x;
  if (idx >= total) return;
  int v    = idx & 7;
  int lane = (idx >> 3) & 31;
  int nt   = (idx >> 8) % ntiles;
  int kt   = (idx >> 8) / ntiles;
  int n = nt * 16 + (lane & 15);
  int k = kt * 32 + (lane >> 4) * 16 + 2 * v;
  u16 lo = fetch_bf(W, k,     n, srcK, srcN, split);
  u16 hi = fetch_bf(W, k + 1, n, srcK, srcN, split);
  dst[idx] = (u32)lo | ((u32)hi << 16);
}

// ---------------------------------------------------------------------------
// nodeFeat (bf16, K padded to 288): [latent[batch[i]] (256) || pos[i] (8) || 0]
// ---------------------------------------------------------------------------
__global__ void build_nodefeat(const float* __restrict__ latent,
                               const float* __restrict__ pos,
                               const int*   __restrict__ batch,
                               u16* __restrict__ nf) {
  int idx = blockIdx.x * blockDim.x + threadIdx.x;
  if (idx >= N_NODES * KP_NODE) return;
  int i = idx / KP_NODE;
  int k = idx - i * KP_NODE;
  float v;
  if (k < 256)      v = latent[(long)batch[i] * 256 + k];
  else if (k < 264) v = pos[(long)i * 8 + (k - 256)];
  else              v = 0.f;
  nf[idx] = f2bf(v);
}

// ---------------------------------------------------------------------------
// WMMA fragment loaders
// ---------------------------------------------------------------------------
__device__ __forceinline__ void load_a_frag(BF16Frag& a, const u16* Abase,
                                            int Kpad, int strip, int kt, int lane) {
  int m = lane & 15, hi = lane >> 4;
  const u32* r = (const u32*)(Abase + (strip * 16 + m) * Kpad + kt * 32 + hi * 16);
  a.u[0] = r[0]; a.u[1] = r[1]; a.u[2] = r[2];  a.u[3] = r[3];
  a.u[4] = r[4]; a.u[5] = r[5]; a.u[6] = r[6]; a.u[7] = r[7];
}

__device__ __forceinline__ void load_b_frag(BF16Frag& b, const u32* Ws,
                                            int kt, int nt, int ntiles, int lane) {
  const uint4* p = (const uint4*)(Ws + ((((kt * ntiles + nt) * 32) + lane) << 3));
  uint4 x = p[0], y = p[1];
  b.u[0] = x.x; b.u[1] = x.y; b.u[2] = x.z; b.u[3] = x.w;
  b.u[4] = y.x; b.u[5] = y.y; b.u[6] = y.z; b.u[7] = y.w;
}

#define WMMA_BF16(a, b, c) \
  __builtin_amdgcn_wmma_f32_16x16x32_bf16(false, (a), false, (b), (short)0, (c), false, false)

// ---------------------------------------------------------------------------
// One GEMM layer over a 64-row LDS tile: H = relu(A @ W + b), bf16 to LDS.
// Each wave owns all STRIPS row strips and NTILES/8 n-tiles; accumulators
// stay live across the whole K loop.
// ---------------------------------------------------------------------------
template <int STRIPS, int NTILES, int KTILES>
__device__ __forceinline__ void gemm_relu_lds(const u16* A, int Kpad,
                                              const u32* Ws, const float* bias,
                                              u16* H, int Hld,
                                              int wave, int lane) {
  constexpr int NTW = (NTILES >= 8) ? (NTILES / 8) : 1;
  if (NTILES < 8 && wave >= NTILES) return;
  int nt0 = (NTILES >= 8) ? wave * NTW : wave;
  int col = lane & 15, hi = lane >> 4;

  v8f acc[STRIPS][NTW];
  #pragma unroll
  for (int j = 0; j < NTW; ++j) {
    float bv = bias[(nt0 + j) * 16 + col];
    #pragma unroll
    for (int s = 0; s < STRIPS; ++s)
      #pragma unroll
      for (int g = 0; g < 8; ++g) acc[s][j][g] = bv;
  }

  for (int kt = 0; kt < KTILES; ++kt) {
    BF16Frag a[STRIPS];
    #pragma unroll
    for (int s = 0; s < STRIPS; ++s) load_a_frag(a[s], A, Kpad, s, kt, lane);
    #pragma unroll
    for (int j = 0; j < NTW; ++j) {
      BF16Frag b;
      load_b_frag(b, Ws, kt, nt0 + j, NTILES, lane);
      #pragma unroll
      for (int s = 0; s < STRIPS; ++s)
        acc[s][j] = WMMA_BF16(a[s].v, b.v, acc[s][j]);
    }
  }

  #pragma unroll
  for (int j = 0; j < NTW; ++j) {
    #pragma unroll
    for (int s = 0; s < STRIPS; ++s) {
      #pragma unroll
      for (int g = 0; g < 8; ++g) {
        float x = acc[s][j][g] > 0.f ? acc[s][j][g] : 0.f;
        int row = s * 16 + g + hi * 8;
        H[row * Hld + (nt0 + j) * 16 + col] = f2bf(x);
      }
    }
  }
}

// Final layer: out = A @ W + b (no relu), f32 store, row/col clamped.
template <int STRIPS, int NTILES, int KTILES>
__device__ __forceinline__ void gemm_out(const u16* A, int Kpad,
                                         const u32* Ws, const float* bias,
                                         float* out, long rowBase, int outCols,
                                         long rowLimit, int wave, int lane) {
  constexpr int NTW = (NTILES >= 8) ? (NTILES / 8) : 1;
  if (NTILES < 8 && wave >= NTILES) return;
  int nt0 = (NTILES >= 8) ? wave * NTW : wave;
  int col = lane & 15, hi = lane >> 4;

  v8f acc[STRIPS][NTW];
  #pragma unroll
  for (int j = 0; j < NTW; ++j) {
    int c = (nt0 + j) * 16 + col;
    float bv = (c < outCols) ? bias[c] : 0.f;
    #pragma unroll
    for (int s = 0; s < STRIPS; ++s)
      #pragma unroll
      for (int g = 0; g < 8; ++g) acc[s][j][g] = bv;
  }

  for (int kt = 0; kt < KTILES; ++kt) {
    BF16Frag a[STRIPS];
    #pragma unroll
    for (int s = 0; s < STRIPS; ++s) load_a_frag(a[s], A, Kpad, s, kt, lane);
    #pragma unroll
    for (int j = 0; j < NTW; ++j) {
      BF16Frag b;
      load_b_frag(b, Ws, kt, nt0 + j, NTILES, lane);
      #pragma unroll
      for (int s = 0; s < STRIPS; ++s)
        acc[s][j] = WMMA_BF16(a[s].v, b.v, acc[s][j]);
    }
  }

  #pragma unroll
  for (int j = 0; j < NTW; ++j) {
    int c = (nt0 + j) * 16 + col;
    if (c < outCols) {
      #pragma unroll
      for (int s = 0; s < STRIPS; ++s) {
        #pragma unroll
        for (int g = 0; g < 8; ++g) {
          long row = rowBase + s * 16 + g + hi * 8;
          if (row < rowLimit) out[row * outCols + c] = acc[s][j][g];
        }
      }
    }
  }
}

// ---------------------------------------------------------------------------
// Node MLP: 64-row tiles, 288->256 relu ->128 relu ->119, fused in LDS.
// ---------------------------------------------------------------------------
__global__ __launch_bounds__(256) void node_mlp(
    const u16* __restrict__ nf,
    const u32* __restrict__ W1s, const float* __restrict__ b1,
    const u32* __restrict__ W2s, const float* __restrict__ b2,
    const u32* __restrict__ W3s, const float* __restrict__ b3,
    float* __restrict__ out) {
  __shared__ u16 sh[64 * KP_NODE + 64 * 256 + 64 * 128];  // 86 KB
  u16* A  = sh;
  u16* H1 = sh + 64 * KP_NODE;
  u16* H2 = H1 + 64 * 256;
  int tid = threadIdx.x, wave = tid >> 5, lane = tid & 31;
  long base = (long)blockIdx.x * 64;

  if (tid == 0) __builtin_prefetch(W1s, 0, 0);  // warm GL2 path

  // stage 64 contiguous node rows (bf16) into LDS; zero-fill past N_NODES
  const u32* src = (const u32*)(nf + base * KP_NODE);
  u32* dstA = (u32*)A;
  int validDw = (int)((N_NODES - base) < 64 ? (N_NODES - base) : 64) * (KP_NODE / 2);
  for (int d = tid; d < 64 * (KP_NODE / 2); d += 256)
    dstA[d] = (d < validDw) ? src[d] : 0u;
  __syncthreads();

  gemm_relu_lds<4, 16, 9>(A,  KP_NODE, W1s, b1, H1, 256, wave, lane);
  __syncthreads();
  gemm_relu_lds<4, 8, 8>(H1, 256, W2s, b2, H2, 128, wave, lane);
  __syncthreads();
  gemm_out<4, 8, 4>(H2, 128, W3s, b3, out, base, 119, N_NODES, wave, lane);
}

// ---------------------------------------------------------------------------
// Edge MLPs: 64-edge tiles; stage [nf[src] || nf[dst]] once (K=576),
// run both the bond-type and bond-dir MLPs off the same A tile.
// ---------------------------------------------------------------------------
__global__ __launch_bounds__(256) void edge_mlp(
    const u16* __restrict__ nf, const int* __restrict__ eidx,
    const u32* __restrict__ tW1s, const float* __restrict__ tb1,
    const u32* __restrict__ tW2s, const float* __restrict__ tb2,
    const u32* __restrict__ tW3s, const float* __restrict__ tb3,
    const u32* __restrict__ dW1s, const float* __restrict__ db1,
    const u32* __restrict__ dW2s, const float* __restrict__ db2,
    const u32* __restrict__ dW3s, const float* __restrict__ db3,
    float* __restrict__ outT, float* __restrict__ outD) {
  __shared__ u16 sh[64 * KP_EDGE + 64 * 256 + 64 * 128];  // 123 KB
  u16* A  = sh;
  u16* H1 = sh + 64 * KP_EDGE;
  u16* H2 = H1 + 64 * 256;
  int tid = threadIdx.x, wave = tid >> 5, lane = tid & 31;
  long base = (long)blockIdx.x * 64;

  if (tid == 0) __builtin_prefetch(tW1s, 0, 0);

  // gather: row r = [nf[src[e]] (144 dw) || nf[dst[e]] (144 dw)]; hits L2
  const u32* nf32 = (const u32*)nf;
  u32* dstA = (u32*)A;
  for (int d = tid; d < 64 * (KP_EDGE / 2); d += 256) {
    int r = d / (KP_EDGE / 2);
    int c = d - r * (KP_EDGE / 2);
    int e = (int)base + r;
    int node = (c < 144) ? eidx[e] : eidx[N_EDGES + e];
    int cc   = (c < 144) ? c : c - 144;
    dstA[d] = nf32[(long)node * 144 + cc];
  }
  __syncthreads();

  // bond-type MLP
  gemm_relu_lds<4, 16, 18>(A,  KP_EDGE, tW1s, tb1, H1, 256, wave, lane);
  __syncthreads();
  gemm_relu_lds<4, 8, 8>(H1, 256, tW2s, tb2, H2, 128, wave, lane);
  __syncthreads();
  gemm_out<4, 1, 4>(H2, 128, tW3s, tb3, outT, base, 4, N_EDGES, wave, lane);
  __syncthreads();
  // bond-dir MLP (reuses A/H1/H2)
  gemm_relu_lds<4, 16, 18>(A,  KP_EDGE, dW1s, db1, H1, 256, wave, lane);
  __syncthreads();
  gemm_relu_lds<4, 8, 8>(H1, 256, dW2s, db2, H2, 128, wave, lane);
  __syncthreads();
  gemm_out<4, 1, 4>(H2, 128, dW3s, db3, outD, base, 3, N_EDGES, wave, lane);
}

// ---------------------------------------------------------------------------
// Launch
// ---------------------------------------------------------------------------
extern "C" void kernel_launch(void* const* d_in, const int* in_sizes, int n_in,
                              void* d_out, int out_size, void* d_ws, size_t ws_size,
                              hipStream_t stream) {
  (void)in_sizes; (void)n_in; (void)out_size; (void)ws_size;
  const float* latent = (const float*)d_in[0];
  const float* pos    = (const float*)d_in[1];
  const int*   batch  = (const int*)d_in[2];
  const int*   eidx   = (const int*)d_in[3];
  const float* nW1 = (const float*)d_in[4];  const float* nb1 = (const float*)d_in[5];
  const float* nW2 = (const float*)d_in[6];  const float* nb2 = (const float*)d_in[7];
  const float* nW3 = (const float*)d_in[8];  const float* nb3 = (const float*)d_in[9];
  const float* tW1 = (const float*)d_in[10]; const float* tb1 = (const float*)d_in[11];
  const float* tW2 = (const float*)d_in[12]; const float* tb2 = (const float*)d_in[13];
  const float* tW3 = (const float*)d_in[14]; const float* tb3 = (const float*)d_in[15];
  const float* dW1 = (const float*)d_in[16]; const float* db1 = (const float*)d_in[17];
  const float* dW2 = (const float*)d_in[18]; const float* db2 = (const float*)d_in[19];
  const float* dW3 = (const float*)d_in[20]; const float* db3 = (const float*)d_in[21];
  float* out = (float*)d_out;

  // workspace layout (all offsets 32B-aligned)
  char* ws = (char*)d_ws;
  long off = 0;
  u16* nf = (u16*)(ws + off); off += (long)N_NODES * KP_NODE * 2;  // 57,600,000
  u32* nW1s = (u32*)(ws + off); off += 9  * 16 * 256 * 4;
  u32* nW2s = (u32*)(ws + off); off += 8  *  8 * 256 * 4;
  u32* nW3s = (u32*)(ws + off); off += 4  *  8 * 256 * 4;
  u32* tW1s = (u32*)(ws + off); off += 18 * 16 * 256 * 4;
  u32* tW2s = (u32*)(ws + off); off += 8  *  8 * 256 * 4;
  u32* tW3s = (u32*)(ws + off); off += 4  *  1 * 256 * 4;
  u32* dW1s = (u32*)(ws + off); off += 18 * 16 * 256 * 4;
  u32* dW2s = (u32*)(ws + off); off += 8  *  8 * 256 * 4;
  u32* dW3s = (u32*)(ws + off); off += 4  *  1 * 256 * 4;

  auto prep = [&](const float* W, u32* dst, int srcK, int srcN,
                  int ktiles, int ntiles, int split) {
    int total = ktiles * ntiles * 256;
    prep_wswz<<<(total + 255) / 256, 256, 0, stream>>>(W, dst, srcK, srcN,
                                                       ntiles, split, total);
  };
  prep(nW1, nW1s, 264, 256,  9, 16, 0);
  prep(nW2, nW2s, 256, 128,  8,  8, 0);
  prep(nW3, nW3s, 128, 119,  4,  8, 0);
  prep(tW1, tW1s, 528, 256, 18, 16, 1);
  prep(tW2, tW2s, 256, 128,  8,  8, 0);
  prep(tW3, tW3s, 128,   4,  4,  1, 0);
  prep(dW1, dW1s, 528, 256, 18, 16, 1);
  prep(dW2, dW2s, 256, 128,  8,  8, 0);
  prep(dW3, dW3s, 128,   3,  4,  1, 0);

  {
    int total = N_NODES * KP_NODE;
    build_nodefeat<<<(total + 255) / 256, 256, 0, stream>>>(latent, pos, batch, nf);
  }

  node_mlp<<<(N_NODES + 63) / 64, 256, 0, stream>>>(nf, nW1s, nb1, nW2s, nb2,
                                                    nW3s, nb3, out);

  edge_mlp<<<N_EDGES / 64, 256, 0, stream>>>(nf, eidx,
                                             tW1s, tb1, tW2s, tb2, tW3s, tb3,
                                             dW1s, db1, dW2s, db2, dW3s, db3,
                                             out + OUT_T, out + OUT_D);
}
